// modi_cgcnn_a2e_46248207843563
// MI455X (gfx1250) — matrix operand order repaired and checked by reading, not compiled, not run on hardware
//
#include <hip/hip_runtime.h>
#include <math.h>

// ---------------- problem constants ----------------
#define N_EDGES   200000
#define N_ANGLES  1600000
#define FDIM      64
#define ADIM      40
#define NCRYS     256
#define KDIM      168      // A + 2F
#define KPAD      192      // padded K for bf16 WMMA (6 steps of 32)
#define NDIM      128      // 2F output cols
#define INV_SQRT_2 0.70710678118654752440f

// ---------------- workspace layout (float units) ----------------
#define OFF_GATED   0LL
#define OFF_SUM1    ((long long)N_ANGLES * NDIM)          // 204,800,000
#define OFF_SQ1     (OFF_SUM1  + (long long)NCRYS * NDIM)
#define OFF_CNT1    (OFF_SQ1   + (long long)NCRYS * NDIM)
#define OFF_MEAN1   (OFF_CNT1  + NCRYS)
#define OFF_RSTD1   (OFF_MEAN1 + (long long)NCRYS * NDIM)
#define OFF_SUM2    (OFF_RSTD1 + (long long)NCRYS * NDIM)
#define OFF_SQ2     (OFF_SUM2  + (long long)NCRYS * FDIM)
#define OFF_CNT2    (OFF_SQ2   + (long long)NCRYS * FDIM)
#define OFF_MEAN2   (OFF_CNT2  + NCRYS)
#define OFF_RSTD2   (OFF_MEAN2 + (long long)NCRYS * FDIM)
#define OFF_SUMMED  (OFF_RSTD2 + (long long)NCRYS * FDIM)
#define OFF_CNTE    (OFF_SUMMED + (long long)N_EDGES * FDIM)
#define OFF_WT      (OFF_CNTE  + N_EDGES)                 // bf16 Wt[128][192] (12288 floats)
#define OFF_END     (OFF_WT    + ((long long)NDIM * KPAD) / 2)

// ---------------- WMMA types ----------------
typedef __bf16 bf16_t;
typedef bf16_t v16bf __attribute__((ext_vector_type(16)));
typedef float  v8f   __attribute__((ext_vector_type(8)));

struct alignas(16) U128 { unsigned int v[4]; };
struct Pack32 { U128 lo, hi; };

__device__ __forceinline__ unsigned short bf16_bits(float f) {
    unsigned int u = __builtin_bit_cast(unsigned int, f);
    unsigned int r = (u + 0x7FFFu + ((u >> 16) & 1u)) >> 16;
    return (unsigned short)r;
}

__device__ __forceinline__ v16bf load_frag(const unsigned short* p0,
                                           const unsigned short* p1) {
    Pack32 t;
    t.lo = *(const U128*)p0;   // 16-byte aligned by construction -> b128 loads
    t.hi = *(const U128*)p1;
    return __builtin_bit_cast(v16bf, t);
}

// ---------------- K0: zero the accumulator region ----------------
__global__ void zero_kernel(float* __restrict__ p, long long n) {
    long long i = (long long)blockIdx.x * blockDim.x + threadIdx.x;
    long long stride = (long long)gridDim.x * blockDim.x;
    for (; i < n; i += stride) p[i] = 0.0f;
}

// ---------------- K0b: W_full(168x128) -> bf16 Wt[n=128][k=192] ----------------
__global__ void prep_w_kernel(const float* __restrict__ W_full,
                              unsigned short* __restrict__ Wt) {
    int idx = blockIdx.x * blockDim.x + threadIdx.x;
    if (idx >= NDIM * KPAD) return;
    int n = idx / KPAD, k = idx % KPAD;
    float v = (k < KDIM) ? W_full[k * NDIM + n] : 0.0f;
    Wt[idx] = bf16_bits(v);
}

// ---------------- K1: gated = [angle_fea | nbr_fea[i0] | nbr_fea[i1]] @ W_full ----
// 128 rows/block, 8 waves; wave w -> M-tile w (16 rows) x all 8 N-tiles.
__global__ __launch_bounds__(256) void gemm_gated_kernel(
    const float* __restrict__ angle_fea,
    const float* __restrict__ nbr_fea,
    const int*   __restrict__ angle_nbr_idx,
    const unsigned short* __restrict__ Wt,   // bf16 [128][192]
    float* __restrict__ gated)               // [N_ANGLES][128]
{
    __shared__ unsigned short sA[128 * 200]; // 16B-aligned rows (200*2=400B)

    const int tid = threadIdx.x;
    const long long blockBase = (long long)blockIdx.x * 128;

    // Stage A tile (128 rows x 192 bf16), gathering the pair features.
    {
        int row   = tid & 127;
        int cbase = (tid >> 7) * 96;
        long long r = blockBase + row;
        const float* af = angle_fea + r * ADIM;
        int i0 = angle_nbr_idx[r * 2 + 0];
        int i1 = angle_nbr_idx[r * 2 + 1];
        const float* n0 = nbr_fea + (long long)i0 * FDIM;
        const float* n1 = nbr_fea + (long long)i1 * FDIM;
        unsigned short* dst = sA + row * 200;
        #pragma unroll 4
        for (int c = cbase; c < cbase + 96; ++c) {
            float v;
            if      (c < ADIM)        v = af[c];
            else if (c < ADIM + FDIM) v = n0[c - ADIM];
            else if (c < KDIM)        v = n1[c - ADIM - FDIM];
            else                      v = 0.0f;
            dst[c] = bf16_bits(v);
        }
    }
    __syncthreads();

    const int lane = tid & 31;
    const int wv   = tid >> 5;
    const int l16  = lane & 15;
    const bool hi  = lane >= 16;
    const unsigned short* aRow = sA + (wv * 16 + l16) * 200;
    const int aOff = hi ? 8 : 0;
    const int bOff = hi ? 8 : 0;
    const v8f vzero = {0.f,0.f,0.f,0.f,0.f,0.f,0.f,0.f};

    #pragma unroll
    for (int nh = 0; nh < 2; ++nh) {
        v8f acc[4];
        #pragma unroll
        for (int nt = 0; nt < 4; ++nt) acc[nt] = vzero;

        #pragma unroll
        for (int k = 0; k < 6; ++k) {
            const int k0 = k * 32;
            // A fragment: lane<16 holds K=[k0..k0+7],[k0+16..k0+23]; lane>=16 +8
            v16bf afrag = load_frag(aRow + k0 + aOff, aRow + k0 + 16 + aOff);
            #pragma unroll
            for (int nt = 0; nt < 4; ++nt) {
                const int Nb = (nh * 4 + nt) * 16;
                // B fragment: lane<16 holds K=[k0..k0+7],[k0+16..k0+23] of col N=Nb+l16; lane>=16 +8
                const unsigned short* bB = Wt + (Nb + l16) * KPAD + k0 + bOff;
                v16bf bfrag = load_frag(bB, bB + 16);
                acc[nt] = __builtin_amdgcn_wmma_f32_16x16x32_bf16(
                    false, afrag, false, bfrag, (short)0, acc[nt], false, false);
            }
        }
        // D layout: lane<16 -> N=l16, M=v; lane>=16 -> M=v+8
        #pragma unroll
        for (int nt = 0; nt < 4; ++nt) {
            const int Nb = (nh * 4 + nt) * 16 + l16;
            #pragma unroll
            for (int v = 0; v < 8; ++v) {
                const int m = v + (hi ? 8 : 0);
                gated[(blockBase + wv * 16 + m) * NDIM + Nb] = acc[nt][v];
            }
        }
    }
}

// ---------------- K2: per-crystal stats over gated (sorted segments) ----------
__global__ __launch_bounds__(128) void angle_stats_kernel(
    const float* __restrict__ gated, const int* __restrict__ seg,
    float* __restrict__ sum1, float* __restrict__ sq1, float* __restrict__ cnt1)
{
    __shared__ int s_seg[256];
    const int t = threadIdx.x;
    const long long base = (long long)blockIdx.x * 256;
    for (int i = t; i < 256; i += 128) s_seg[i] = seg[base + i];
    __syncthreads();

    float sm = 0.f, sq = 0.f, rc = 0.f;
    int cur = s_seg[0];
    for (int r = 0; r < 256; ++r) {
        int s = s_seg[r];
        if (s != cur) {
            atomicAdd(&sum1[cur * NDIM + t], sm);
            atomicAdd(&sq1 [cur * NDIM + t], sq);
            if (t == 0) atomicAdd(&cnt1[cur], rc);
            sm = 0.f; sq = 0.f; rc = 0.f; cur = s;
        }
        float x = gated[(base + r) * NDIM + t];
        sm += x; sq += x * x; rc += 1.f;
    }
    atomicAdd(&sum1[cur * NDIM + t], sm);
    atomicAdd(&sq1 [cur * NDIM + t], sq);
    if (t == 0) atomicAdd(&cnt1[cur], rc);
}

// ---------------- K3/K6: finalize mean / rstd ----------------
__global__ void finalize_stats_kernel(
    const float* __restrict__ sum, const float* __restrict__ sq,
    const float* __restrict__ cnt, float* __restrict__ mean,
    float* __restrict__ rstd, int ncol, int total)
{
    int i = blockIdx.x * blockDim.x + threadIdx.x;
    if (i >= total) return;
    float c = fmaxf(cnt[i / ncol], 1.0f);
    float m = sum[i] / c;
    float v = sq[i] / c - m * m;
    mean[i] = m;
    rstd[i] = rsqrtf(fmaxf(v, 0.0f) + 1e-5f);
}

// ---------------- K4: norm + relu/tanh gate + scatter to edges --------------
__global__ __launch_bounds__(256) void gate_scatter_kernel(
    const float* __restrict__ gated, const int* __restrict__ angle_nbr_idx,
    const int* __restrict__ segA,
    const float* __restrict__ mean1, const float* __restrict__ rstd1,
    const float* __restrict__ g1, const float* __restrict__ b1,
    const float* __restrict__ Wm,
    float* __restrict__ summed, float* __restrict__ cntE)
{
    const int tid = threadIdx.x;
    const int lane = tid & 31, wv = tid >> 5;
    const long long row = (long long)blockIdx.x * 8 + wv;
    const int sg = segA[row];
    const float* gr = gated + row * NDIM;
    const float* mu = mean1 + sg * NDIM;
    const float* rs = rstd1 + sg * NDIM;
    const int c0 = lane, c1 = lane + 32, c2 = lane + 64, c3 = lane + 96;

    float core0 = fmaxf((gr[c0] - mu[c0]) * rs[c0] * g1[c0] + b1[c0], 0.f);
    float core1 = fmaxf((gr[c1] - mu[c1]) * rs[c1] * g1[c1] + b1[c1], 0.f);
    float f0    = (gr[c2] - mu[c2]) * rs[c2] * g1[c2] + b1[c2];
    float f1    = (gr[c3] - mu[c3]) * rs[c3] * g1[c3] + b1[c3];

    float p = f0 * Wm[lane] + f1 * Wm[lane + 32];
    #pragma unroll
    for (int off = 16; off > 0; off >>= 1) p += __shfl_xor(p, off, 32);
    float s = tanhf(p);

    const int src = angle_nbr_idx[row * 2];
    atomicAdd(&summed[(long long)src * FDIM + lane],      s * core0);
    atomicAdd(&summed[(long long)src * FDIM + lane + 32], s * core1);
    if (lane == 0) atomicAdd(&cntE[src], 1.0f);
}

// ---------------- K5: normalize by count (in place) + per-crystal edge stats -
__global__ __launch_bounds__(64) void edge_stats_kernel(
    float* __restrict__ summed, const float* __restrict__ cntE,
    const int* __restrict__ seg,
    float* __restrict__ sum2, float* __restrict__ sq2, float* __restrict__ cnt2)
{
    __shared__ int   s_seg[250];
    __shared__ float s_inv[250];
    const int t = threadIdx.x;
    const long long base = (long long)blockIdx.x * 250;
    for (int i = t; i < 250; i += 64) {
        s_seg[i] = seg[base + i];
        s_inv[i] = 1.0f / fmaxf(cntE[base + i], 1.0f);
    }
    __syncthreads();

    float sm = 0.f, sq = 0.f, rc = 0.f;
    int cur = s_seg[0];
    for (int r = 0; r < 250; ++r) {
        int s = s_seg[r];
        if (s != cur) {
            atomicAdd(&sum2[cur * FDIM + t], sm);
            atomicAdd(&sq2 [cur * FDIM + t], sq);
            if (t == 0) atomicAdd(&cnt2[cur], rc);
            sm = 0.f; sq = 0.f; rc = 0.f; cur = s;
        }
        float x = summed[(base + r) * FDIM + t] * s_inv[r];
        summed[(base + r) * FDIM + t] = x;   // nbr_sumed in place
        sm += x; sq += x * x; rc += 1.f;
    }
    atomicAdd(&sum2[cur * FDIM + t], sm);
    atomicAdd(&sq2 [cur * FDIM + t], sq);
    if (t == 0) atomicAdd(&cnt2[cur], rc);
}

// ---------------- K7: crystal_norm2 + 2 residual MLPs + final relu ----------
__global__ __launch_bounds__(256) void residual_out_kernel(
    const float* __restrict__ ns, const int* __restrict__ segE,
    const float* __restrict__ mean2, const float* __restrict__ rstd2,
    const float* __restrict__ g2, const float* __restrict__ b2,
    const float* __restrict__ W1a, const float* __restrict__ b1a,
    const float* __restrict__ W2a, const float* __restrict__ b2a,
    const float* __restrict__ W1b, const float* __restrict__ b1b,
    const float* __restrict__ W2b, const float* __restrict__ b2b,
    const float* __restrict__ nbr_fea, float* __restrict__ out)
{
    __shared__ float h_sh[8][64];
    __shared__ float t_sh[8][32];
    const int tid = threadIdx.x;
    const int lane = tid & 31, wv = tid >> 5;
    const long long row = (long long)blockIdx.x * 8 + wv;
    const int sg = segE[row];
    const int c0 = lane, c1 = lane + 32;

    float h0 = (ns[row*FDIM+c0] - mean2[sg*FDIM+c0]) * rstd2[sg*FDIM+c0] * g2[c0] + b2[c0];
    float h1 = (ns[row*FDIM+c1] - mean2[sg*FDIM+c1]) * rstd2[sg*FDIM+c1] * g2[c1] + b2[c1];
    h_sh[wv][c0] = h0; h_sh[wv][c1] = h1;
    __syncthreads();

    // MLP a: 64 -> 32 -> 64
    float tj = b1a[lane];
    #pragma unroll 8
    for (int c = 0; c < 64; ++c) tj += h_sh[wv][c] * W1a[c * 32 + lane];
    t_sh[wv][lane] = fmaxf(tj, 0.f);
    __syncthreads();
    float a0 = b2a[c0], a1 = b2a[c1];
    #pragma unroll 8
    for (int j = 0; j < 32; ++j) {
        float tv = t_sh[wv][j];
        a0 += tv * W2a[j * 64 + c0];
        a1 += tv * W2a[j * 64 + c1];
    }
    h0 += a0; h1 += a1;
    __syncthreads();
    h_sh[wv][c0] = h0; h_sh[wv][c1] = h1;
    __syncthreads();

    // MLP b: 64 -> 32 -> 64
    float uj = b1b[lane];
    #pragma unroll 8
    for (int c = 0; c < 64; ++c) uj += h_sh[wv][c] * W1b[c * 32 + lane];
    __syncthreads();
    t_sh[wv][lane] = fmaxf(uj, 0.f);
    __syncthreads();
    float d0 = b2b[c0], d1 = b2b[c1];
    #pragma unroll 8
    for (int j = 0; j < 32; ++j) {
        float tv = t_sh[wv][j];
        d0 += tv * W2b[j * 64 + c0];
        d1 += tv * W2b[j * 64 + c1];
    }
    h0 += d0; h1 += d1;

    out[row*FDIM+c0] = INV_SQRT_2 * fmaxf(nbr_fea[row*FDIM+c0] + h0, 0.f);
    out[row*FDIM+c1] = INV_SQRT_2 * fmaxf(nbr_fea[row*FDIM+c1] + h1, 0.f);
}

// ---------------- launch ----------------
extern "C" void kernel_launch(void* const* d_in, const int* in_sizes, int n_in,
                              void* d_out, int out_size, void* d_ws, size_t ws_size,
                              hipStream_t stream) {
    (void)in_sizes; (void)n_in; (void)out_size; (void)ws_size;

    const float* nbr_fea       = (const float*)d_in[0];
    const float* angle_fea     = (const float*)d_in[1];
    const int*   angle_nbr_idx = (const int*)  d_in[2];
    const int*   segE          = (const int*)  d_in[3];
    const int*   segA          = (const int*)  d_in[4];
    const float* W_full        = (const float*)d_in[5];
    const float* W_mask        = (const float*)d_in[6];
    const float* g1            = (const float*)d_in[7];
    const float* b1            = (const float*)d_in[8];
    const float* g2            = (const float*)d_in[9];
    const float* b2            = (const float*)d_in[10];
    const float* W1a = (const float*)d_in[11];
    const float* b1a = (const float*)d_in[12];
    const float* W2a = (const float*)d_in[13];
    const float* b2a = (const float*)d_in[14];
    const float* W1b = (const float*)d_in[15];
    const float* b1b = (const float*)d_in[16];
    const float* W2b = (const float*)d_in[17];
    const float* b2b = (const float*)d_in[18];
    float* out = (float*)d_out;

    float* ws = (float*)d_ws;
    float* gated  = ws + OFF_GATED;
    float* sum1   = ws + OFF_SUM1;
    float* sq1    = ws + OFF_SQ1;
    float* cnt1   = ws + OFF_CNT1;
    float* mean1  = ws + OFF_MEAN1;
    float* rstd1  = ws + OFF_RSTD1;
    float* sum2   = ws + OFF_SUM2;
    float* sq2    = ws + OFF_SQ2;
    float* cnt2   = ws + OFF_CNT2;
    float* mean2  = ws + OFF_MEAN2;
    float* rstd2  = ws + OFF_RSTD2;
    float* summed = ws + OFF_SUMMED;
    float* cntE   = ws + OFF_CNTE;
    unsigned short* Wt = (unsigned short*)(ws + OFF_WT);

    // zero accumulators (atomically-added regions + counters)
    zero_kernel<<<2048, 256, 0, stream>>>(ws + OFF_SUM1, OFF_WT - OFF_SUM1);

    // W -> bf16 transposed
    prep_w_kernel<<<(NDIM * KPAD + 255) / 256, 256, 0, stream>>>(W_full, Wt);

    // main WMMA GEMM + gather
    gemm_gated_kernel<<<N_ANGLES / 128, 256, 0, stream>>>(
        angle_fea, nbr_fea, angle_nbr_idx, Wt, gated);

    // crystal norm #1 stats
    angle_stats_kernel<<<N_ANGLES / 256, 128, 0, stream>>>(gated, segA, sum1, sq1, cnt1);
    finalize_stats_kernel<<<(NCRYS * NDIM + 255) / 256, 256, 0, stream>>>(
        sum1, sq1, cnt1, mean1, rstd1, NDIM, NCRYS * NDIM);

    // gate + scatter to edges
    gate_scatter_kernel<<<N_ANGLES / 8, 256, 0, stream>>>(
        gated, angle_nbr_idx, segA, mean1, rstd1, g1, b1, W_mask, summed, cntE);

    // mean over incident angles + crystal norm #2 stats
    edge_stats_kernel<<<N_EDGES / 250, 64, 0, stream>>>(summed, cntE, segE, sum2, sq2, cnt2);
    finalize_stats_kernel<<<(NCRYS * FDIM + 255) / 256, 256, 0, stream>>>(
        sum2, sq2, cnt2, mean2, rstd2, FDIM, NCRYS * FDIM);

    // norm #2 + residual MLPs + final relu
    residual_out_kernel<<<N_EDGES / 8, 256, 0, stream>>>(
        summed, segE, mean2, rstd2, g2, b2,
        W1a, b1a, W2a, b2a, W1b, b1b, W2b, b2b, nbr_fea, out);
}
